// FlattenDeflatten_59889023976201
// MI455X (gfx1250) — matrix operand order repaired and checked by reading, not compiled, not run on hardware
//
#include <hip/hip_runtime.h>
#include <math.h>

typedef __attribute__((ext_vector_type(16))) _Float16 v16h;
typedef __attribute__((ext_vector_type(8)))  _Float16 v8h;
typedef __attribute__((ext_vector_type(8)))  float    v8f;

#define BB   32
#define CC   128
#define BCHN 32
#define DHN  256
#define NN   14400

#define TWD  128   // LDS tile width (positions), incl. 4+4 halo
#define TN   120   // valid outputs per tile (NN / TN = 120 tiles exactly)
#define NT   120
#define RP1  136   // row pitch (halfs) for seq  (CC  + 8)
#define RP2  40    // row pitch for flat          (BCH + 8)
#define RP3  264   // row pitch for hid           (DH  + 8)

// ---------------------------------------------------------------------------
// Weight reorder: wA[co * (Cin*3) + k*Cin + ci] = (f16) w[co][ci][k]
// With t = k*Cin + ci and 32 | Cin, every 32-wide WMMA K-chunk has a single
// tap k, so B-fragments are contiguous LDS reads.
// ---------------------------------------------------------------------------
__global__ void prep_weights_fdf(const float* __restrict__ wf,
                                 const float* __restrict__ w1,
                                 const float* __restrict__ w2,
                                 _Float16* __restrict__ A1,
                                 _Float16* __restrict__ A2,
                                 _Float16* __restrict__ A3)
{
  int tid = blockIdx.x * blockDim.x + threadIdx.x;
  const int S1 = BCHN * CC * 3;          // 12288
  const int S2 = DHN * BCHN * 3;         // 24576
  const int S3 = CC * DHN * 3;           // 98304
  if (tid < S1) {
    int co = tid / (CC * 3), t = tid - co * (CC * 3);
    int k = t >> 7, ci = t & (CC - 1);
    A1[tid] = (_Float16)wf[(co * CC + ci) * 3 + k];
  } else if (tid < S1 + S2) {
    int i = tid - S1;
    int co = i / (BCHN * 3), t = i - co * (BCHN * 3);
    int k = t >> 5, ci = t & (BCHN - 1);
    A2[i] = (_Float16)w1[(co * BCHN + ci) * 3 + k];
  } else if (tid < S1 + S2 + S3) {
    int i = tid - S1 - S2;
    int co = i / (DHN * 3), t = i - co * (DHN * 3);
    int k = t >> 8, ci = t & (DHN - 1);
    A3[i] = (_Float16)w2[(co * DHN + ci) * 3 + k];
  }
}

// ---------------------------------------------------------------------------
// One 16(M) x 16(N) output tile of conv1d-as-GEMM, K = Cin*3, via
// v_wmma_f32_16x16x32_f16. lds_in is [pos][ci] (pitch RPIN halfs).
// A frag: lane m = lane%16, halfs e   -> t=32ks+16h+e.
// B frag: lane n = lane%16, halfs e   -> t=32ks+16h+e; k=t/Cin, ci=t%Cin,
//         read lds_in[(p0+n+k-1)*RPIN + ci] (contiguous 16 halfs).
// ---------------------------------------------------------------------------
template<int CIN, int RPIN>
__device__ __forceinline__ v8f conv_tile(const _Float16* __restrict__ lds_in,
                                         const _Float16* __restrict__ wA,
                                         const float* __restrict__ bias,
                                         int mt, int p0, int h, int n)
{
  constexpr int KD = CIN * 3;
  v8f acc;
#pragma unroll
  for (int r = 0; r < 8; ++r) acc[r] = bias[mt * 16 + r + 8 * h];
  const _Float16* wrow = wA + (size_t)(mt * 16 + n) * KD;
#pragma unroll
  for (int ks = 0; ks < KD / 32; ++ks) {
    const int tbB = ks * 32 + 16 * h;
    const int k   = tbB / CIN;       // compile-time per ks (32 | CIN)
    const int cib = tbB % CIN;       // multiple of 16 -> 32B aligned
    const int row = p0 + n + k - 1;  // may dip into halo/garbage: by design
    v16h bf, af;
    {
      const v8h* bp = (const v8h*)(lds_in + row * RPIN + cib);
      v8h lo = bp[0], hi = bp[1];
#pragma unroll
      for (int i = 0; i < 8; ++i) { bf[i] = lo[i]; bf[i + 8] = hi[i]; }
    }
    {
      const int tbA = ks * 32 + 16 * h;
      v8h lo = *(const v8h*)(wrow + tbA);
      v8h hi = *(const v8h*)(wrow + tbA + 8);
#pragma unroll
      for (int i = 0; i < 8; ++i) { af[i] = lo[i]; af[i + 8] = hi[i]; }
    }
    acc = __builtin_amdgcn_wmma_f32_16x16x32_f16(false, af, false, bf,
                                                 (short)0, acc, false, false);
  }
  return acc;
}

// ---------------------------------------------------------------------------
// Fused: gather -> conv1 -> conv2+GELU -> conv3+embed -> scatter.
// grid = (NT, BB), block = 256 (8 waves, wave w owns column tile p0 = 16w).
// ---------------------------------------------------------------------------
__global__ __launch_bounds__(256)
void fused_fdf(const float* __restrict__ latent,
               const float* __restrict__ b_flat,
               const float* __restrict__ b_d1,
               const float* __restrict__ b_d2,
               const float* __restrict__ embed,
               const int*   __restrict__ walk,
               const _Float16* __restrict__ wA1,
               const _Float16* __restrict__ wA2,
               const _Float16* __restrict__ wA3,
               float* __restrict__ out_recon,
               float* __restrict__ out_flat)
{
  __shared__ _Float16 lds_seq [TWD * RP1];
  __shared__ _Float16 lds_flat[TWD * RP2];
  __shared__ _Float16 lds_hid [TWD * RP3];

  const int tile = blockIdx.x;
  const int b    = blockIdx.y;
  const int n0   = tile * TN;

  // ---- Stage A: gather latent along Hilbert walk, cvt f32->f16 into LDS ---
  for (int idx = threadIdx.x; idx < CC * TWD; idx += 256) {
    int ci = idx >> 7;           // TWD == 128
    int p  = idx & (TWD - 1);
    int g  = n0 - 4 + p;
    float v = 0.0f;
    if (g >= 0 && g < NN)
      v = latent[(size_t)b * CC * NN + (size_t)ci * NN + walk[g]];
    lds_seq[p * RP1 + ci] = (_Float16)v;
  }
  __syncthreads();

  const int lane = threadIdx.x & 31;
  const int wv   = threadIdx.x >> 5;
  const int h    = lane >> 4;
  const int n    = lane & 15;
  const int p0   = wv * 16;
  const int p    = p0 + n;                       // output position (LDS coord)
  const int g    = n0 - 4 + p;                   // global sequence position
  const bool valid = (p >= 4) && (p < 4 + TN) && (g < NN);

  // ---- Stage B: conv1 (128 -> 32), emit flat (f32 global) + LDS f16 -------
#pragma unroll
  for (int mt = 0; mt < BCHN / 16; ++mt) {
    v8f acc = conv_tile<CC, RP1>(lds_seq, wA1, b_flat, mt, p0, h, n);
#pragma unroll
    for (int r = 0; r < 8; ++r) {
      int co = mt * 16 + r + 8 * h;
      lds_flat[p * RP2 + co] = (_Float16)acc[r];
      if (valid)
        out_flat[(size_t)b * BCHN * NN + (size_t)co * NN + g] = acc[r];
    }
  }
  __syncthreads();

  // ---- Stage C: conv2 (32 -> 256) + exact GELU, into LDS f16 --------------
#pragma unroll
  for (int mt = 0; mt < DHN / 16; ++mt) {
    v8f acc = conv_tile<BCHN, RP2>(lds_flat, wA2, b_d1, mt, p0, h, n);
#pragma unroll
    for (int r = 0; r < 8; ++r) {
      float x  = acc[r];
      float gx = 0.5f * x * (1.0f + erff(x * 0.70710678118654752f));
      lds_hid[p * RP3 + mt * 16 + r + 8 * h] = (_Float16)gx;
    }
  }
  __syncthreads();

  // ---- Stage D: conv3 (256 -> 128) + spatial_embed, scatter via walk ------
  const int hw = valid ? walk[g] : 0;
#pragma unroll
  for (int mt = 0; mt < CC / 16; ++mt) {
    v8f acc = conv_tile<DHN, RP3>(lds_hid, wA3, b_d2, mt, p0, h, n);
#pragma unroll
    for (int r = 0; r < 8; ++r) {
      int c = mt * 16 + r + 8 * h;
      if (valid) {
        float v = acc[r] + embed[(size_t)c * NN + g];
        out_recon[(size_t)b * CC * NN + (size_t)c * NN + hw] = v;
      }
    }
  }
}

// ---------------------------------------------------------------------------
extern "C" void kernel_launch(void* const* d_in, const int* in_sizes, int n_in,
                              void* d_out, int out_size, void* d_ws, size_t ws_size,
                              hipStream_t stream) {
  const float* latent = (const float*)d_in[0];
  const float* w_flat = (const float*)d_in[1];
  const float* b_flat = (const float*)d_in[2];
  const float* w_d1   = (const float*)d_in[3];
  const float* b_d1   = (const float*)d_in[4];
  const float* w_d2   = (const float*)d_in[5];
  const float* b_d2   = (const float*)d_in[6];
  const float* embed  = (const float*)d_in[7];
  const int*   walk   = (const int*)d_in[8];

  _Float16* wA1 = (_Float16*)d_ws;               // 32*384   halfs
  _Float16* wA2 = wA1 + BCHN * CC * 3;           // 256*96   halfs
  _Float16* wA3 = wA2 + DHN * BCHN * 3;          // 128*768  halfs

  float* out_recon = (float*)d_out;
  float* out_flat  = out_recon + (size_t)BB * CC * NN;

  const int wtot = BCHN * CC * 3 + DHN * BCHN * 3 + CC * DHN * 3;
  prep_weights_fdf<<<(wtot + 255) / 256, 256, 0, stream>>>(
      w_flat, w_d1, w_d2, wA1, wA2, wA3);

  dim3 grid(NT, BB);
  fused_fdf<<<grid, 256, 0, stream>>>(latent, b_flat, b_d1, b_d2, embed, walk,
                                      wA1, wA2, wA3, out_recon, out_flat);
}
